// SelfAttentionBlock_71494025609283
// MI455X (gfx1250) — matrix operand order repaired and checked by reading, not compiled, not run on hardware
//
#include <hip/hip_runtime.h>
#include <hip/hip_bf16.h>

// ---------------- types ----------------
typedef __attribute__((ext_vector_type(16))) __bf16 v16bf;
typedef __attribute__((ext_vector_type(8)))  __bf16 v8bf;
typedef __attribute__((ext_vector_type(8)))  float  v8f;

static __device__ __forceinline__ unsigned short f2bfbits(float f) {
  unsigned int u = __float_as_uint(f);
  unsigned int r = (u + 0x7FFFu + ((u >> 16) & 1u)) >> 16;
  return (unsigned short)r;
}

// A-operand (16x32 bf16): lane<16 holds row M=lane, K 0-15; lane>=16 row M=lane-16, K 16-31.
static __device__ __forceinline__ v16bf load_pair16(const unsigned short* p0,
                                                    const unsigned short* p1) {
  v16bf r;
  *reinterpret_cast<v8bf*>(&r)       = *reinterpret_cast<const v8bf*>(p0);
  *(reinterpret_cast<v8bf*>(&r) + 1) = *reinterpret_cast<const v8bf*>(p1);
  return r;
}
// B-operand (32x16 bf16): lane holds one column, 16 consecutive K values -> 32 contiguous bytes.
static __device__ __forceinline__ v16bf load_b16(const unsigned short* p) {
  return load_pair16(p, p + 8);
}

static __device__ __forceinline__ v8f wmma_bf16(v16bf a, v16bf b, v8f c) {
  return __builtin_amdgcn_wmma_f32_16x16x32_bf16(false, a, false, b, (short)0, c,
                                                 false, false);
}

#define BATCH 4
#define CCH   256
#define NPIX  4096

// ---------------- pass 0: convert / transpose ----------------
__global__ void cvt_kernel(const float* __restrict__ src,
                           unsigned short* __restrict__ dst, int n) {
  int i = blockIdx.x * blockDim.x + threadIdx.x;
  if (i < n) dst[i] = f2bfbits(src[i]);
}

// LDS-tiled transpose: x[b][c][n] f32  ->  xT[b][n][c] bf16 (coalesced both sides)
__global__ void __launch_bounds__(256)
xpose_kernel(const float* __restrict__ x, unsigned short* __restrict__ xT) {
  __shared__ float tile[32][33];
  const int bid = blockIdx.x;              // 4 * 8 * 128 = 4096 blocks
  const int n0 = (bid & 127) * 32;
  const int c0 = ((bid >> 7) & 7) * 32;
  const int b  = bid >> 10;
  const int tx = threadIdx.x & 31;
  const int ty = threadIdx.x >> 5;
#pragma unroll
  for (int k = 0; k < 4; ++k) {
    int c = ty + 8 * k;
    tile[c][tx] = x[((size_t)(b * CCH + c0 + c)) * NPIX + n0 + tx];
  }
  __syncthreads();
#pragma unroll
  for (int k = 0; k < 4; ++k) {
    int n = ty + 8 * k;
    xT[((size_t)(b * NPIX + n0 + n)) * CCH + c0 + tx] = f2bfbits(tile[tx][n]);
  }
}

// ---------------- pass 1: Q/K/V projections via WMMA (software-pipelined) ----------------
// blocks [0,1024):   (b, ntile)          -> Qt[b][n][32], Kt[b][n][32]
// blocks [1024,5120): (b, cgroup, ntile) -> V[b][c][n]
__global__ void __launch_bounds__(32)
proj_kernel(const unsigned short* __restrict__ xT,
            const unsigned short* __restrict__ Wqb,
            const unsigned short* __restrict__ Wkb,
            const unsigned short* __restrict__ Wvb,
            const float* __restrict__ bq, const float* __restrict__ bk,
            const float* __restrict__ bv,
            unsigned short* __restrict__ Qt, unsigned short* __restrict__ Kt,
            unsigned short* __restrict__ Vv) {
  const int id = blockIdx.x;
  const int lane = threadIdx.x;
  const int lanecol = lane & 15;
  const int lanehalf = lane >> 4;
  const int aoff = lanehalf ? 16 : 0;
  const v8f z8 = {0.f, 0.f, 0.f, 0.f, 0.f, 0.f, 0.f, 0.f};

  if (id < 1024) {
    const int b = id >> 8, ntile = id & 255;
    const unsigned short* xrow =
        xT + ((size_t)(b * NPIX + ntile * 16 + lanecol)) * CCH;
    auto loadA = [&](int cc) {
      return load_pair16(xrow + cc * 32 + aoff, xrow + cc * 32 + aoff + 8);
    };
    auto loadW = [&](const unsigned short* W, int qt, int cc) {
      return load_b16(W + (qt * 16 + lanecol) * CCH + cc * 32 + lanehalf * 16);
    };
    v8f accq[2] = {z8, z8};
    v8f acck[2] = {z8, z8};
    // prologue: chunk 0 operands in flight
    v16bf aC   = loadA(0);
    v16bf wqC0 = loadW(Wqb, 0, 0), wqC1 = loadW(Wqb, 1, 0);
    v16bf wkC0 = loadW(Wkb, 0, 0), wkC1 = loadW(Wkb, 1, 0);
#pragma unroll
    for (int cc = 0; cc < 8; ++cc) {
      v16bf aN, wqN0, wqN1, wkN0, wkN1;
      if (cc < 7) {  // prefetch next chunk before consuming current
        aN   = loadA(cc + 1);
        wqN0 = loadW(Wqb, 0, cc + 1);
        wqN1 = loadW(Wqb, 1, cc + 1);
        wkN0 = loadW(Wkb, 0, cc + 1);
        wkN1 = loadW(Wkb, 1, cc + 1);
      }
      accq[0] = wmma_bf16(aC, wqC0, accq[0]);
      accq[1] = wmma_bf16(aC, wqC1, accq[1]);
      acck[0] = wmma_bf16(aC, wkC0, acck[0]);
      acck[1] = wmma_bf16(aC, wkC1, acck[1]);
      if (cc < 7) {
        aC = aN; wqC0 = wqN0; wqC1 = wqN1; wkC0 = wkN0; wkC1 = wkN1;
      }
    }
#pragma unroll
    for (int qt = 0; qt < 2; ++qt) {
      int q = qt * 16 + lanecol;
      float bqv = bq[q], bkv = bk[q];
#pragma unroll
      for (int r = 0; r < 8; ++r) {
        int m = r + 8 * lanehalf;
        size_t o = ((size_t)(b * NPIX) + ntile * 16 + m) * 32 + q;
        Qt[o] = f2bfbits(accq[qt][r] + bqv);
        Kt[o] = f2bfbits(acck[qt][r] + bkv);
      }
    }
  } else {
    const int vid = id - 1024;
    const int b = vid >> 10;
    const int cgroup = (vid >> 8) & 3;
    const int ntile = vid & 255;
    const int n = ntile * 16 + lanecol;
    const unsigned short* xrow = xT + ((size_t)(b * NPIX + n)) * CCH;
    auto loadBx = [&](int cc) {
      return load_b16(xrow + cc * 32 + lanehalf * 16);
    };
    auto loadWv = [&](int t, int cc) {
      const unsigned short* wr =
          Wvb + ((cgroup * 4 + t) * 16 + lanecol) * CCH + cc * 32 + aoff;
      return load_pair16(wr, wr + 8);
    };
    v8f acc[4] = {z8, z8, z8, z8};
    v16bf bxC = loadBx(0);
    v16bf aC0 = loadWv(0, 0), aC1 = loadWv(1, 0);
    v16bf aC2 = loadWv(2, 0), aC3 = loadWv(3, 0);
#pragma unroll
    for (int cc = 0; cc < 8; ++cc) {
      v16bf bxN, aN0, aN1, aN2, aN3;
      if (cc < 7) {  // prefetch next chunk before consuming current
        bxN = loadBx(cc + 1);
        aN0 = loadWv(0, cc + 1);
        aN1 = loadWv(1, cc + 1);
        aN2 = loadWv(2, cc + 1);
        aN3 = loadWv(3, cc + 1);
      }
      acc[0] = wmma_bf16(aC0, bxC, acc[0]);
      acc[1] = wmma_bf16(aC1, bxC, acc[1]);
      acc[2] = wmma_bf16(aC2, bxC, acc[2]);
      acc[3] = wmma_bf16(aC3, bxC, acc[3]);
      if (cc < 7) {
        bxC = bxN; aC0 = aN0; aC1 = aN1; aC2 = aN2; aC3 = aN3;
      }
    }
#pragma unroll
    for (int t = 0; t < 4; ++t) {
#pragma unroll
      for (int r = 0; r < 8; ++r) {
        int cout = (cgroup * 4 + t) * 16 + r + 8 * lanehalf;
        size_t o = ((size_t)(b * CCH + cout)) * NPIX + n;
        Vv[o] = f2bfbits(acc[t][r] + bv[cout]);
      }
    }
  }
}

// ---------------- pass 2: flash attention via WMMA + async LDS staging ----------------
// block = 256 threads = 8 waves: 4 i-tiles x 2 c-halves. grid = B * 64 = 256 blocks.
// Per 32-j chunk: stage K (2KB) + V (16KB) into LDS with global_load_async_to_lds,
// double-buffered (ASYNCcnt pipeline), all 8 waves consume from LDS.

// exactly 5 async ops per lane per chunk (uniform per wave): 4x b128 (V) + 1x b64 (K)
static __device__ __forceinline__ void issue_chunk(
    const unsigned short* Vbase, const unsigned short* Kbase,
    unsigned vlds_base, unsigned klds_base, unsigned jb, unsigned t) {
  unsigned voffg = (t * (unsigned)NPIX + jb) * 2u;   // V row c = t, 64B per row
  unsigned vlds = vlds_base + t * 64u;
#pragma unroll
  for (unsigned p = 0; p < 4; ++p) {
    asm volatile("global_load_async_to_lds_b128 %0, %1, %2"
                 :
                 : "v"(vlds + p * 16u), "v"(voffg + p * 16u), "s"(Vbase)
                 : "memory");
  }
  unsigned j = t >> 3, part = t & 7;                 // K row j, 8B per thread
  unsigned koffg = (jb + j) * 64u + part * 8u;
  unsigned klds = klds_base + j * 64u + part * 8u;
  asm volatile("global_load_async_to_lds_b64 %0, %1, %2"
               :
               : "v"(klds), "v"(koffg), "s"(Kbase)
               : "memory");
}

__global__ void __launch_bounds__(256)
attn_kernel(const unsigned short* __restrict__ Qt,
            const unsigned short* __restrict__ Kt,
            const unsigned short* __restrict__ Vv,
            const float* __restrict__ x, const float* __restrict__ gamma,
            float* __restrict__ out) {
  __shared__ __align__(16) unsigned short Kbuf[2][32 * 32];   // [j][q]
  __shared__ __align__(16) unsigned short Vbuf[2][256 * 32];  // [c][j]
  __shared__ __align__(16) unsigned short Pls[8][16 * 32];    // per-wave P tile

  const int b = blockIdx.x >> 6;
  const int igroup = blockIdx.x & 63;
  const int tid = threadIdx.x;
  const int wave = tid >> 5;
  const int lane = tid & 31;
  const int lanecol = lane & 15;
  const int lanehalf = lane >> 4;
  const int aoff = lanehalf ? 16 : 0;
  const int chalf = wave & 1;
  const int itile = igroup * 4 + (wave >> 1);
  const float scale = 0.0625f;  // 1/sqrt(256)
  const v8f z8 = {0.f, 0.f, 0.f, 0.f, 0.f, 0.f, 0.f, 0.f};

  const unsigned short* Vbase = Vv + (size_t)b * CCH * NPIX;
  const unsigned short* Kbase = Kt + (size_t)b * NPIX * 32;

  // Q tile as A operand (K reduction dim is exactly 32)
  const unsigned short* qrow =
      Qt + ((size_t)(b * NPIX) + itile * 16 + lanecol) * 32;
  v16bf qa = load_pair16(qrow + aoff, qrow + aoff + 8);

  v8f acc[8];
  float m_r[8], l_r[8];
#pragma unroll
  for (int t = 0; t < 8; ++t) acc[t] = z8;
#pragma unroll
  for (int r = 0; r < 8; ++r) { m_r[r] = -1e30f; l_r[r] = 0.f; }

  const unsigned vb0 = (unsigned)(size_t)&Vbuf[0][0];
  const unsigned vb1 = (unsigned)(size_t)&Vbuf[1][0];
  const unsigned kb0a = (unsigned)(size_t)&Kbuf[0][0];
  const unsigned kb1a = (unsigned)(size_t)&Kbuf[1][0];

  issue_chunk(Vbase, Kbase, vb0, kb0a, 0u, (unsigned)tid);

  for (int k = 0; k < NPIX / 32; ++k) {
    const int cur = k & 1;
    const unsigned jb = (unsigned)(k * 32);
    if (k + 1 < NPIX / 32) {
      issue_chunk(Vbase, Kbase, cur ? vb0 : vb1, cur ? kb0a : kb1a,
                  jb + 32u, (unsigned)tid);
      asm volatile("s_wait_asynccnt 5" ::: "memory");  // prev chunk resident
    } else {
      asm volatile("s_wait_asynccnt 0" ::: "memory");
    }
    __syncthreads();

    const unsigned short* Kc = &Kbuf[cur][0];
    const unsigned short* Vc = &Vbuf[cur][0];

    // scores: 2 WMMAs (j chunk of 32)
    v16bf kb_lo = load_b16(Kc + lanecol * 32 + lanehalf * 16);
    v16bf kb_hi = load_b16(Kc + (16 + lanecol) * 32 + lanehalf * 16);
    v8f s0 = wmma_bf16(qa, kb_lo, z8);
    v8f s1 = wmma_bf16(qa, kb_hi, z8);

    float alpha[8];
    unsigned short* pw = &Pls[wave][0];
#pragma unroll
    for (int r = 0; r < 8; ++r) {
      float a0 = s0[r] * scale, a1 = s1[r] * scale;
      float t = fmaxf(a0, a1);
      t = fmaxf(t, __shfl_xor(t, 1));
      t = fmaxf(t, __shfl_xor(t, 2));
      t = fmaxf(t, __shfl_xor(t, 4));
      t = fmaxf(t, __shfl_xor(t, 8));  // row reduce within 16-lane half
      float mn = fmaxf(m_r[r], t);
      alpha[r] = __expf(m_r[r] - mn);
      m_r[r] = mn;
      float p0 = __expf(a0 - mn);
      float p1 = __expf(a1 - mn);
      float rs = p0 + p1;
      rs += __shfl_xor(rs, 1);
      rs += __shfl_xor(rs, 2);
      rs += __shfl_xor(rs, 4);
      rs += __shfl_xor(rs, 8);
      l_r[r] = l_r[r] * alpha[r] + rs;
      int m = r + 8 * lanehalf;  // D-layout row for this lane half
      pw[m * 32 + lanecol] = f2bfbits(p0);
      pw[m * 32 + 16 + lanecol] = f2bfbits(p1);
    }
    // re-read P in A-operand layout (per-wave region; DS ops in-order per wave)
    const unsigned short* prow = pw + lanecol * 32;
    v16bf pa = load_pair16(prow + aoff, prow + aoff + 8);

    // preload all 8 V operands from LDS, then WMMA chain (waits amortize,
    // acc rescale VALU co-executes with XDL WMMAs)
    v16bf vbr[8];
#pragma unroll
    for (int t8 = 0; t8 < 8; ++t8) {
      int c = chalf * 128 + t8 * 16 + lanecol;
      vbr[t8] = load_b16(Vc + c * 32 + lanehalf * 16);
    }
#pragma unroll
    for (int t8 = 0; t8 < 8; ++t8) {
#pragma unroll
      for (int r = 0; r < 8; ++r) acc[t8][r] *= alpha[r];
      acc[t8] = wmma_bf16(pa, vbr[t8], acc[t8]);
    }
    __syncthreads();  // all waves done with buf[cur] before it is re-filled
  }

  const float g = gamma[0];
#pragma unroll
  for (int r = 0; r < 8; ++r) l_r[r] = 1.0f / l_r[r];
#pragma unroll
  for (int t8 = 0; t8 < 8; ++t8) {
    int c = chalf * 128 + t8 * 16 + lanecol;
    // per lane: fixed c, 8 consecutive i -> two float4 stores
    size_t o0 = ((size_t)(b * CCH + c)) * NPIX + itile * 16 + 8 * lanehalf;
    float4 xv0 = *reinterpret_cast<const float4*>(x + o0);
    float4 xv1 = *reinterpret_cast<const float4*>(x + o0 + 4);
    float4 r0, r1;
    r0.x = g * (acc[t8][0] * l_r[0]) + xv0.x;
    r0.y = g * (acc[t8][1] * l_r[1]) + xv0.y;
    r0.z = g * (acc[t8][2] * l_r[2]) + xv0.z;
    r0.w = g * (acc[t8][3] * l_r[3]) + xv0.w;
    r1.x = g * (acc[t8][4] * l_r[4]) + xv1.x;
    r1.y = g * (acc[t8][5] * l_r[5]) + xv1.y;
    r1.z = g * (acc[t8][6] * l_r[6]) + xv1.z;
    r1.w = g * (acc[t8][7] * l_r[7]) + xv1.w;
    *reinterpret_cast<float4*>(out + o0) = r0;
    *reinterpret_cast<float4*>(out + o0 + 4) = r1;
  }
}

// ---------------- launcher ----------------
extern "C" void kernel_launch(void* const* d_in, const int* in_sizes, int n_in,
                              void* d_out, int out_size, void* d_ws,
                              size_t ws_size, hipStream_t stream) {
  const float* x     = (const float*)d_in[0];
  const float* Wq    = (const float*)d_in[1];
  const float* bq    = (const float*)d_in[2];
  const float* Wk    = (const float*)d_in[3];
  const float* bk    = (const float*)d_in[4];
  const float* Wv    = (const float*)d_in[5];
  const float* bv    = (const float*)d_in[6];
  const float* gamma = (const float*)d_in[7];
  float* out = (float*)d_out;

  // workspace carve (bf16 elements, all 16B aligned)
  unsigned short* xT  = (unsigned short*)d_ws;            // B*N*C = 4,194,304
  unsigned short* WqB = xT + (size_t)BATCH * NPIX * CCH;  // 8192
  unsigned short* WkB = WqB + 32 * CCH;                   // 8192
  unsigned short* WvB = WkB + 32 * CCH;                   // 65536
  unsigned short* Qt  = WvB + CCH * CCH;                  // B*N*32
  unsigned short* Kt  = Qt + (size_t)BATCH * NPIX * 32;
  unsigned short* Vv  = Kt + (size_t)BATCH * NPIX * 32;   // B*C*N

  xpose_kernel<<<BATCH * (CCH / 32) * (NPIX / 32), 256, 0, stream>>>(x, xT);
  cvt_kernel<<<32, 256, 0, stream>>>(Wq, WqB, 32 * CCH);
  cvt_kernel<<<32, 256, 0, stream>>>(Wk, WkB, 32 * CCH);
  cvt_kernel<<<256, 256, 0, stream>>>(Wv, WvB, CCH * CCH);
  proj_kernel<<<5120, 32, 0, stream>>>(xT, WqB, WkB, WvB, bq, bk, bv, Qt, Kt, Vv);
  attn_kernel<<<BATCH * 64, 256, 0, stream>>>(Qt, Kt, Vv, x, gamma, out);
}
